// Model_45183055954583
// MI455X (gfx1250) — matrix operand order repaired and checked by reading, not compiled, not run on hardware
//
#include <hip/hip_runtime.h>

// ---------------------------------------------------------------------------
// CDNA5 (gfx1250, wave32) GNN forward: encoder MLP -> 5x SAGEConv (+BN/ReLU)
// -> decoder MLP.  Activations stored bf16 (halves L2 gather traffic, feeds
// WMMA A-operand directly); all accumulation in f32 via
// v_wmma_f32_16x16x32_bf16.
// ---------------------------------------------------------------------------

typedef __attribute__((ext_vector_type(16))) __bf16 v16bf;
typedef __attribute__((ext_vector_type(8)))  __bf16 bf8v;
typedef __attribute__((ext_vector_type(4)))  __bf16 bf4v;
typedef __attribute__((ext_vector_type(8)))  float  v8f;
typedef __attribute__((ext_vector_type(4)))  float  f4;

#define HDIM 128
#define EPSV 1e-5f

// ------------------------------- utility kernels ---------------------------
__global__ void k_zero_i32(int* p, int n) {
  int i = blockIdx.x * 256 + threadIdx.x;
  if (i < n) p[i] = 0;
}
__global__ void k_zero_f32(float* p, int n) {
  int i = blockIdx.x * 256 + threadIdx.x;
  if (i < n) p[i] = 0.0f;
}
__global__ void k_copy_i32(const int* a, int* b, int n) {
  int i = blockIdx.x * 256 + threadIdx.x;
  if (i < n) b[i] = a[i];
}

// degree: atomicAdd over edge destinations
__global__ void k_deg(const int* __restrict__ dst, int* __restrict__ deg, int e) {
  int i = blockIdx.x * 256 + threadIdx.x;
  if (i < e) atomicAdd(&deg[dst[i]], 1);
}

// two-level exclusive scan over degrees -> CSR row_start
__global__ void k_scan1(const int* __restrict__ deg, int* __restrict__ rs,
                        int* __restrict__ blks, int n) {
  __shared__ int sm[256];
  int t = threadIdx.x;
  int i = blockIdx.x * 256 + t;
  int v = (i < n) ? deg[i] : 0;
  sm[t] = v;
  __syncthreads();
  for (int off = 1; off < 256; off <<= 1) {
    int x = sm[t];
    int y = (t >= off) ? sm[t - off] : 0;
    __syncthreads();
    sm[t] = x + y;
    __syncthreads();
  }
  if (i < n) rs[i] = sm[t] - v;              // exclusive within block
  if (t == 255) blks[blockIdx.x] = sm[255];  // block total
}
__global__ void k_scan2(int* blks, int nb) {
  __shared__ int sm[256];
  int t = threadIdx.x;
  int v = (t < nb) ? blks[t] : 0;
  sm[t] = v;
  __syncthreads();
  for (int off = 1; off < 256; off <<= 1) {
    int x = sm[t];
    int y = (t >= off) ? sm[t - off] : 0;
    __syncthreads();
    sm[t] = x + y;
    __syncthreads();
  }
  if (t < nb) blks[t] = sm[t] - v;           // exclusive block offsets
}
__global__ void k_scan3(int* rs, const int* blks, int n) {
  int i = blockIdx.x * 256 + threadIdx.x;
  if (i < n) rs[i] += blks[blockIdx.x];
}

// CSR fill: slot = cursor[dst]++ ; csr_src[slot] = src
__global__ void k_fill(const int* __restrict__ src, const int* __restrict__ dst,
                       int* __restrict__ cursor, int* __restrict__ csr, int e) {
  int i = blockIdx.x * 256 + threadIdx.x;
  if (i < e) {
    int slot = atomicAdd(&cursor[dst[i]], 1);
    csr[slot] = src[i];
  }
}
__global__ void k_deginv(const int* __restrict__ deg, float* __restrict__ dinv, int n) {
  int i = blockIdx.x * 256 + threadIdx.x;
  if (i < n) dinv[i] = 1.0f / (float)max(deg[i], 1);
}

// ------------------------------- encoder layer 1 ---------------------------
// hid[n, j] = relu( sum_k concat(x,fx)[n,k] * W1[k,j] + b1[j] ),  K=8, J=256
__global__ void k_enc1(const float* __restrict__ x, const float* __restrict__ fx,
                       const float* __restrict__ W1, const float* __restrict__ b1,
                       __bf16* __restrict__ hid, int n) {
  __shared__ float in8[8];
  int nid = blockIdx.x;
  int t = threadIdx.x;
  if (t < 3) in8[t] = x[(size_t)nid * 3 + t];
  else if (t < 8) in8[t] = fx[(size_t)nid * 5 + (t - 3)];
  __syncthreads();
  float s = b1[t];
#pragma unroll
  for (int k = 0; k < 8; ++k) s += in8[k] * W1[k * 256 + t];
  hid[(size_t)nid * 256 + t] = (__bf16)fmaxf(s, 0.0f);
}

// ------------------------------- weight packing ----------------------------
// Pack f32 [Ksrc, M] (rows koff..koff+Ksrc-1 of a K_tot-row stack) into the
// wave32 WMMA B-operand layout:  element((n_tile*nk + kblk)*32 + lane, j)
// holds B[kblk*32 + (lane>=16?8:0) + (j&7) + (j>=8?16:0)][n_tile*16 + (lane&15)].
__global__ void k_pack(const float* __restrict__ src, __bf16* __restrict__ dstp,
                       int Ktot, int Ksrc, int M, int koff) {
  int i = blockIdx.x * 256 + threadIdx.x;
  if (i >= Ksrc * M) return;
  int ks = i / M, c = i % M;
  int k = koff + ks;
  int kblk = k >> 5;
  int w = k & 31;
  int lane = (c & 15) + (((w >> 3) & 1) ? 16 : 0);
  int j = (w & 7) + ((w >= 16) ? 8 : 0);
  size_t pos = ((((size_t)(c >> 4) * (Ktot >> 5) + kblk) * 32) + lane) * 16 + j;
  dstp[pos] = (__bf16)src[(size_t)ks * M + c];
}

// ------------------------------- aggregation -------------------------------
// one wave32 per node; lane owns 4 channels; coalesced 256B row gathers (L2)
// z rows are bf16; accumulate in f32; write mean back as bf16.
__global__ void k_agg(const __bf16* __restrict__ z, const int* __restrict__ csr,
                      const int* __restrict__ rs, const int* __restrict__ deg,
                      const float* __restrict__ dinv, __bf16* __restrict__ agg, int n) {
  int t = threadIdx.x;
  int node = blockIdx.x * 8 + (t >> 5);
  int lane = t & 31;
  if (node >= n) return;
  int start = rs[node];
  int cnt = deg[node];
  f4 acc = {0.f, 0.f, 0.f, 0.f};
  const __bf16* zc = z + (size_t)lane * 4;
  for (int i = 0; i < cnt; ++i) {
    int s = csr[start + i];
    bf4v v = *(const bf4v*)(zc + (size_t)s * HDIM);
    acc += __builtin_convertvector(v, f4);
  }
  float d = dinv[node];
  *(bf4v*)(agg + (size_t)node * HDIM + lane * 4) =
      __builtin_convertvector(acc * d, bf4v);
}

// ------------------------------- WMMA GEMM ---------------------------------
// C[r, by*128 + c] = act( sum_k A[r,k] * W[k,c] + bias[c] ),  A/B/C bf16.
// A columns 0..K0-1 from A0 (stride lda0), K0.. from A1 (stride lda1).
// 256 threads = 8 waves; wave owns a 16-row x 128-col stripe (8 v8f accums).
// Optionally accumulates per-channel f32 sum / sum-of-squares for BatchNorm.
__global__ void __launch_bounds__(256)
k_gemm(const __bf16* __restrict__ A0, int lda0,
       const __bf16* __restrict__ A1, int lda1, int K0split, int K,
       const __bf16* __restrict__ Bp, const float* __restrict__ bias,
       __bf16* __restrict__ C, int ldc, int nrows, int relu, int stats,
       float* __restrict__ bn_sum, float* __restrict__ bn_sq) {
  __shared__ float s_sum[128];
  __shared__ float s_sq[128];
  const int t = threadIdx.x;
  if (stats && t < 128) { s_sum[t] = 0.f; s_sq[t] = 0.f; }
  __syncthreads();

  const int lane = t & 31;
  const int wave = t >> 5;
  const int r0 = (blockIdx.x * 8 + wave) * 16;
  const int by = blockIdx.y;
  const int nk = K >> 5;
  const bool active = (r0 < nrows);  // wave-uniform; EXEC stays all-ones

  if (active) {
    const int row = r0 + (lane & 15);        // nrows % 16 == 0 for this model
    const int kbase = (lane < 16) ? 0 : 8;   // A-operand K sub-block per half-wave
    v8f acc[8];
    const v8f vz = {0.f, 0.f, 0.f, 0.f, 0.f, 0.f, 0.f, 0.f};
#pragma unroll
    for (int tt = 0; tt < 8; ++tt) acc[tt] = vz;

    for (int ksb = 0; ksb < nk; ++ksb) {
      const int k0 = ksb << 5;
      const __bf16* base;
      int col;
      if (A1 && k0 >= K0split) { base = A1 + (size_t)row * lda1; col = k0 - K0split + kbase; }
      else                     { base = A0 + (size_t)row * lda0; col = k0 + kbase; }
      // A fragment: two b128 loads, register-concatenated (no cvt needed)
      bf8v c0 = *(const bf8v*)(base + col);        // K = k0+kbase .. +7
      bf8v c1 = *(const bf8v*)(base + col + 16);   // K = k0+kbase+16 .. +23
      v16bf a;
#pragma unroll
      for (int j = 0; j < 8; ++j) { a[j] = c0[j]; a[8 + j] = c1[j]; }

      const __bf16* bptr = Bp + ((((size_t)(by * 8) * nk + ksb) * 32 + lane) << 4);
#pragma unroll
      for (int tt = 0; tt < 8; ++tt) {
        v16bf b = *(const v16bf*)(bptr + (size_t)tt * nk * 512);
        acc[tt] = __builtin_amdgcn_wmma_f32_16x16x32_bf16(
            false, a, false, b, (short)0, acc[tt], false, false);
      }
    }

    const int rowoff = (lane < 16) ? 0 : 8;
#pragma unroll
    for (int tt = 0; tt < 8; ++tt) {
      const int col = by * 128 + tt * 16 + (lane & 15);
      const float bv = bias ? bias[col] : 0.0f;
      float lsum = 0.f, lsq = 0.f;
#pragma unroll
      for (int v = 0; v < 8; ++v) {
        float val = acc[tt][v] + bv;
        if (relu) val = fmaxf(val, 0.0f);
        C[(size_t)(r0 + v + rowoff) * ldc + col] = (__bf16)val;
        lsum += val;
        lsq += val * val;
      }
      if (stats) {
        atomicAdd(&s_sum[tt * 16 + (lane & 15)], lsum);  // ds_add_f32
        atomicAdd(&s_sq[tt * 16 + (lane & 15)], lsq);
      }
    }
  }

  if (stats) {
    __syncthreads();
    if (t < 128) {
      atomicAdd(&bn_sum[t], s_sum[t]);
      atomicAdd(&bn_sq[t], s_sq[t]);
    }
  }
}

// ------------------------------- batch norm --------------------------------
__global__ void k_bnfin(const float* __restrict__ bsum, const float* __restrict__ bsq,
                        const float* __restrict__ g, const float* __restrict__ b,
                        float* __restrict__ scale, float* __restrict__ shift, float invN) {
  int t = threadIdx.x;
  if (t < HDIM) {
    float m = bsum[t] * invN;
    float v = bsq[t] * invN - m * m;
    v = fmaxf(v, 0.0f);
    float sc = g[t] * rsqrtf(v + EPSV);
    scale[t] = sc;
    shift[t] = b[t] - m * sc;
  }
}
__global__ void k_bnrelu(__bf16* __restrict__ z, const float* __restrict__ scale,
                         const float* __restrict__ shift, int total) {
  int i = blockIdx.x * 256 + threadIdx.x;
  if (i < total) {
    int c = i & (HDIM - 1);
    float v = (float)z[i] * scale[c] + shift[c];
    z[i] = (__bf16)fmaxf(v, 0.0f);
  }
}

// ------------------------------- decoder layer 2 ---------------------------
// out[n, j] = sum_k hid[n,k] * W2[k,j] + b2[j]   (K=256, J=4), hid bf16
__global__ void k_dec2(const __bf16* __restrict__ hid, const float* __restrict__ W2,
                       const float* __restrict__ b2, float* __restrict__ out, int n) {
  int i = blockIdx.x * 256 + threadIdx.x;
  if (i >= n * 4) return;
  int nid = i >> 2, j = i & 3;
  float s = b2[j];
  const bf8v* hp = (const bf8v*)(hid + (size_t)nid * 256);
#pragma unroll 4
  for (int kk = 0; kk < 32; ++kk) {
    bf8v h = hp[kk];
    int k = kk * 8;
#pragma unroll
    for (int u = 0; u < 8; ++u) s += (float)h[u] * W2[(k + u) * 4 + j];
  }
  out[i] = s;
}

// ------------------------------- host launcher -----------------------------
static inline char* carve(char** p, size_t bytes) {
  char* r = *p;
  *p += (bytes + 255) & ~(size_t)255;
  return r;
}
static inline int cdiv(int a, int b) { return (a + b - 1) / b; }

extern "C" void kernel_launch(void* const* d_in, const int* in_sizes, int n_in,
                              void* d_out, int out_size, void* d_ws, size_t ws_size,
                              hipStream_t stream) {
  const float* x  = (const float*)d_in[0];
  const float* fx = (const float*)d_in[1];
  const int* geo  = (const int*)d_in[2];
  const int N = in_sizes[0] / 3;
  const int E = in_sizes[2] / 2;
  const int* src = geo;
  const int* dst = geo + E;

  const float* encW1 = (const float*)d_in[3];
  const float* encb1 = (const float*)d_in[4];
  const float* encW2 = (const float*)d_in[5];
  const float* encb2 = (const float*)d_in[6];
  const float *sageWl[5], *sagebl[5], *sageWr[5];
  for (int i = 0; i < 5; ++i) {
    sageWl[i] = (const float*)d_in[7 + 3 * i];
    sagebl[i] = (const float*)d_in[8 + 3 * i];
    sageWr[i] = (const float*)d_in[9 + 3 * i];
  }
  const float *bng[4], *bnb[4];
  for (int i = 0; i < 4; ++i) {
    bng[i] = (const float*)d_in[22 + 2 * i];
    bnb[i] = (const float*)d_in[23 + 2 * i];
  }
  const float* decW1 = (const float*)d_in[30];
  const float* decb1 = (const float*)d_in[31];
  const float* decW2 = (const float*)d_in[32];
  const float* decb2 = (const float*)d_in[33];

  // workspace carve (hid aliases agg's phase: agg = hid base, disjoint in time)
  char* p = (char*)d_ws;
  __bf16* zA  = (__bf16*)carve(&p, (size_t)N * HDIM * 2);
  __bf16* zB  = (__bf16*)carve(&p, (size_t)N * HDIM * 2);
  __bf16* hid = (__bf16*)carve(&p, (size_t)N * 256 * 2);
  __bf16* agg = hid;  // reused as agg during SAGE layers
  int* deg_i    = (int*)carve(&p, (size_t)N * 4);
  float* dinv   = (float*)carve(&p, (size_t)N * 4);
  int* row_st   = (int*)carve(&p, (size_t)N * 4);
  int* cursor   = (int*)carve(&p, (size_t)N * 4);
  int* blks     = (int*)carve(&p, 256 * 4);
  int* csr      = (int*)carve(&p, (size_t)E * 4);
  float* bn_sum = (float*)carve(&p, HDIM * 4);
  float* bn_sq  = (float*)carve(&p, HDIM * 4);
  float* bn_sc  = (float*)carve(&p, HDIM * 4);
  float* bn_sh  = (float*)carve(&p, HDIM * 4);
  __bf16* pw    = (__bf16*)carve(&p, (size_t)7 * 32768 * 2);
  __bf16* pw_enc2 = pw;                // 256x128
  __bf16* pw_sage[5];
  for (int i = 0; i < 5; ++i) pw_sage[i] = pw + (size_t)32768 * (1 + i);  // 256x128 each
  __bf16* pw_dec1 = pw + (size_t)32768 * 6;  // 128x256

  const int nbN = cdiv(N, 256);
  const int nbE = cdiv(E, 256);

  // ---- CSR build (deterministic up to FP summation order) ----
  k_zero_i32<<<nbN, 256, 0, stream>>>(deg_i, N);
  k_deg<<<nbE, 256, 0, stream>>>(dst, deg_i, E);
  k_scan1<<<nbN, 256, 0, stream>>>(deg_i, row_st, blks, N);
  k_scan2<<<1, 256, 0, stream>>>(blks, nbN);
  k_scan3<<<nbN, 256, 0, stream>>>(row_st, blks, N);
  k_copy_i32<<<nbN, 256, 0, stream>>>(row_st, cursor, N);
  k_fill<<<nbE, 256, 0, stream>>>(src, dst, cursor, csr, E);
  k_deginv<<<nbN, 256, 0, stream>>>(deg_i, dinv, N);

  // ---- pack weights to bf16 WMMA-B layout ----
  k_pack<<<cdiv(256 * 128, 256), 256, 0, stream>>>(encW2, pw_enc2, 256, 256, 128, 0);
  for (int i = 0; i < 5; ++i) {
    k_pack<<<cdiv(128 * 128, 256), 256, 0, stream>>>(sageWl[i], pw_sage[i], 256, 128, 128, 0);
    k_pack<<<cdiv(128 * 128, 256), 256, 0, stream>>>(sageWr[i], pw_sage[i], 256, 128, 128, 128);
  }
  k_pack<<<cdiv(128 * 256, 256), 256, 0, stream>>>(decW1, pw_dec1, 128, 128, 256, 0);

  // ---- encoder ----
  k_enc1<<<N, 256, 0, stream>>>(x, fx, encW1, encb1, hid, N);
  dim3 g1(cdiv(N, 128), 1);
  k_gemm<<<g1, 256, 0, stream>>>(hid, 256, nullptr, 0, 1 << 30, 256,
                                 pw_enc2, encb2, zA, HDIM, N, 0, 0, nullptr, nullptr);

  // ---- SAGE layers ----
  __bf16* cur = zA;
  __bf16* oth = zB;
  for (int l = 0; l < 5; ++l) {
    k_agg<<<cdiv(N, 8), 256, 0, stream>>>(cur, csr, row_st, deg_i, dinv, agg, N);
    int st = (l < 4);
    if (st) {
      k_zero_f32<<<1, 256, 0, stream>>>(bn_sum, HDIM);
      k_zero_f32<<<1, 256, 0, stream>>>(bn_sq, HDIM);
    }
    k_gemm<<<g1, 256, 0, stream>>>(agg, HDIM, cur, HDIM, 128, 256,
                                   pw_sage[l], sagebl[l], oth, HDIM, N, 0, st, bn_sum, bn_sq);
    if (st) {
      k_bnfin<<<1, 128, 0, stream>>>(bn_sum, bn_sq, bng[l], bnb[l], bn_sc, bn_sh, 1.0f / (float)N);
      k_bnrelu<<<cdiv(N * HDIM, 256), 256, 0, stream>>>(oth, bn_sc, bn_sh, N * HDIM);
    }
    __bf16* tmp = cur; cur = oth; oth = tmp;
  }

  // ---- decoder ----
  dim3 g2(cdiv(N, 128), 2);
  k_gemm<<<g2, 256, 0, stream>>>(cur, HDIM, nullptr, 0, 1 << 30, 128,
                                 pw_dec1, decb1, hid, 256, N, 1, 0, nullptr, nullptr);
  k_dec2<<<cdiv(N * 4, 256), 256, 0, stream>>>(hid, decW2, decb2, (float*)d_out, N);

  (void)n_in; (void)out_size; (void)ws_size;
}
